// GaussianAtt_35631048687753
// MI455X (gfx1250) — matrix operand order repaired and checked by reading, not compiled, not run on hardware
//
#include <hip/hip_runtime.h>
#include <hip/hip_bf16.h>

// ---------------------------------------------------------------------------
// Problem constants (from the reference)
// ---------------------------------------------------------------------------
#define BB   8
#define HH   96
#define WW   96
#define CCH  256
#define NHEAD 8
#define HDIM 32
#define NTOK (BB * HH * WW)            // 73728 tokens
#define NELT ((size_t)NTOK * CCH)      // 18,874,368 elements
static const float KSCALE = 0.17677669529663687f;   // 32^-0.5

typedef __attribute__((ext_vector_type(16))) __bf16 bhalf16;
typedef __attribute__((ext_vector_type(8)))  __bf16 bhalf8;
typedef __attribute__((ext_vector_type(8)))  float  floatx8;
typedef __attribute__((ext_vector_type(4)))  float  floatx4;

// ---------------------------------------------------------------------------
// WMMA helper — CDNA5 v_wmma_f32_16x16x32_bf16
// ---------------------------------------------------------------------------
__device__ inline floatx8 wmma_bf16(bhalf16 a, bhalf16 b, floatx8 c) {
    return __builtin_amdgcn_wmma_f32_16x16x32_bf16(
        false, a, false, b, (short)0, c, false, false);
}

__device__ inline bhalf16 join8(bhalf8 lo, bhalf8 hi) {
    return __builtin_shufflevector(lo, hi, 0, 1, 2, 3, 4, 5, 6, 7,
                                           8, 9, 10, 11, 12, 13, 14, 15);
}

// A fragment 16x32 (MxK) from bf16 row-major LDS; chunks are 16B aligned
// (requires ld*2 % 16 == 0).  lane: m = lane&15, k-base = 16*(lane>>4).
__device__ inline bhalf16 frag_a_lds(const __bf16* p, int ld) {
    const int lane = threadIdx.x & 31;
    const __bf16* row = p + (lane & 15) * ld + ((lane >> 4) * 16);
    bhalf8 lo = *(const bhalf8*)(row);
    bhalf8 hi = *(const bhalf8*)(row + 8);
    return join8(lo, hi);
}

// A fragment from f32 row-major LDS (scores), converted in registers.
__device__ inline bhalf16 frag_a_f32(const float* p, int ld) {
    const int lane = threadIdx.x & 31;
    const float* row = p + (lane & 15) * ld + ((lane >> 4) * 16);
    floatx4 a0 = *(const floatx4*)(row);
    floatx4 a1 = *(const floatx4*)(row + 4);
    floatx4 a2 = *(const floatx4*)(row + 8);
    floatx4 a3 = *(const floatx4*)(row + 12);
    bhalf16 a;
#pragma unroll
    for (int i = 0; i < 4; ++i) {
        a[i]      = (__bf16)a0[i];
        a[i + 4]  = (__bf16)a1[i];
        a[i + 8]  = (__bf16)a2[i];
        a[i + 12] = (__bf16)a3[i];
    }
    return a;
}

// B fragment where the 16 k-elements for a lane are contiguous 32B in LDS
// (used for Q*K^T on row-major K, and for A*V on transposed V).
// Requires ld*2 % 32 == 0 and 32B-aligned base.
__device__ inline bhalf16 frag_row32(const __bf16* p, int ld) {
    const int lane = threadIdx.x & 31;
    return *(const bhalf16*)(p + (lane & 15) * ld + ((lane >> 4) * 16));
}

// B fragment from the pre-swizzled global weight buffer: one 32B vector load.
__device__ inline bhalf16 frag_b_swz(const __bf16* p) {
    return *(const bhalf16*)(p + (threadIdx.x & 31) * 16);
}

// ---------------------------------------------------------------------------
// Kernel 0: pre-swizzle Wq|Wk|Wv (48 n-tiles) and Wo (16 n-tiles) into bf16
// B-fragment layout: dst[((ks*NT + nt)*32 + lane)*16 + i] = W[k, n]
//   with n = nt*16 + (lane&15), k = ks*32 + (lane>>4)*16 + i.
// ---------------------------------------------------------------------------
__global__ __launch_bounds__(256) void wprep_kernel(
    const float* __restrict__ Wq, const float* __restrict__ Wk,
    const float* __restrict__ Wv, const float* __restrict__ Wo,
    __bf16* __restrict__ swzQKV, __bf16* __restrict__ swzO)
{
    const int idx = blockIdx.x * 256 + threadIdx.x;   // < 262144
    const int i    = idx & 15;
    const int lane = (idx >> 4) & 31;
    const int rest = idx >> 9;
    if (rest < 48 * 8) {            // QKV region
        const int nt = rest % 48;
        const int ks = rest / 48;
        const int mat = nt >> 4;
        const float* W = (mat == 0) ? Wq : (mat == 1) ? Wk : Wv;
        const int n = (nt & 15) * 16 + (lane & 15);
        const int k = ks * 32 + (lane >> 4) * 16 + i;
        swzQKV[(size_t)idx] = (__bf16)W[(size_t)k * CCH + n];
    } else {                         // Wo region
        const int r2 = rest - 48 * 8;
        const int nt = r2 % 16;
        const int ks = r2 / 16;
        const int n = nt * 16 + (lane & 15);
        const int k = ks * 32 + (lane >> 4) * 16 + i;
        swzO[(size_t)(((ks * 16 + nt) * 32 + lane) * 16 + i)] =
            (__bf16)Wo[(size_t)k * CCH + n];
    }
}

// ---------------------------------------------------------------------------
// Kernel 1: fused q/k/v projection (A-fragments hoisted; swizzled weights)
// ---------------------------------------------------------------------------
#define ASTR 264   // padded LDS row stride (bf16), rows 16B-aligned (528 B)

__global__ __launch_bounds__(256) void qkv_kernel(
    const float* __restrict__ x,  const float* __restrict__ enc,
    const __bf16* __restrict__ swzQKV,
    const float* __restrict__ bq, const float* __restrict__ bk,
    const float* __restrict__ bv,
    __bf16* __restrict__ qb, __bf16* __restrict__ kb, __bf16* __restrict__ vb)
{
    __shared__ __align__(32) __bf16 As[16 * ASTR];

    const int tid = threadIdx.x;
    const int m0  = blockIdx.x * 16;

    for (int idx = tid; idx < 16 * 256; idx += 256) {
        const int r = idx >> 8, c = idx & 255;
        As[r * ASTR + c] = (__bf16)x[(size_t)(m0 + r) * CCH + c];
    }
    __syncthreads();

    const int wave = tid >> 5;
    const int lane = tid & 31;
    const int nloc = lane & 15;
    const int mh   = (lane >> 4) * 8;

    bhalf16 afr[8];
#pragma unroll
    for (int ks = 0; ks < 8; ++ks) afr[ks] = frag_a_lds(As + ks * 32, ASTR);

    for (int nt = wave; nt < 48; nt += 8) {
        const int mat = nt >> 4;            // 0=q, 1=k, 2=v
        const int n0  = (nt & 15) * 16;
        const float* bm = (mat == 0) ? bq : (mat == 1) ? bk : bv;

        floatx8 acc = {};
#pragma unroll
        for (int ks = 0; ks < 8; ++ks) {
            if (ks < 7)
                __builtin_prefetch(swzQKV + (size_t)((ks + 1) * 48 + nt) * 512
                                          + lane * 16, 0, 0);
            bhalf16 b = frag_b_swz(swzQKV + (size_t)(ks * 48 + nt) * 512);
            acc = wmma_bf16(afr[ks], b, acc);
        }

        const int col = n0 + nloc;
        const float bias = bm[col];
#pragma unroll
        for (int r = 0; r < 8; ++r) {
            const size_t g = (size_t)(m0 + mh + r) * CCH + col;
            float v = acc[r] + bias;
            if (mat == 0)      qb[g] = (__bf16)v;
            else if (mat == 1) kb[g] = (__bf16)((v + enc[g]) * KSCALE);
            else               vb[g] = (__bf16)(v + enc[g]);
        }
    }
}

// ---------------------------------------------------------------------------
// Kernel 2: depthwise 5x5 LePE conv on v (pad 2). One thread per output.
// ---------------------------------------------------------------------------
__global__ __launch_bounds__(256) void lepe_kernel(
    const __bf16* __restrict__ vb, const float* __restrict__ cw,
    const float* __restrict__ cb, float* __restrict__ lepe)
{
    const size_t idx = (size_t)blockIdx.x * 256 + threadIdx.x;
    if (idx >= NELT) return;
    const int c = (int)(idx & 255);
    const size_t pix = idx >> 8;
    const int w = (int)(pix % WW);
    const int h = (int)((pix / WW) % HH);
    const int b = (int)(pix / (WW * HH));

    float acc = cb[c];
#pragma unroll
    for (int kh = 0; kh < 5; ++kh) {
        const int ih = h + kh - 2;
        if (ih < 0 || ih >= HH) continue;
#pragma unroll
        for (int kw = 0; kw < 5; ++kw) {
            const int iw = w + kw - 2;
            if (iw < 0 || iw >= WW) continue;
            const size_t g = (((size_t)b * HH + ih) * WW + iw) * CCH + c;
            acc += (float)vb[g] * cw[(kh * 5 + kw) * CCH + c];
        }
    }
    lepe[idx] = acc;
}

// ---------------------------------------------------------------------------
// Kernels 3/4: axial attention, one block per (batch, line, head).
// MODE 0: width-axis  (seq = w),  writes bf16 v1
// MODE 1: height-axis (seq = h),  writes f32 out + lepe
// LDS: Q 96x48(bf16) + K 96x48(bf16) + V^T 32x112(bf16) + S 96x100(f32)
//      = 9216 + 9216 + 7168 + 38400 = 64,000 B
// ---------------------------------------------------------------------------
#define QSTR 48    // 96 B rows  -> 32B-aligned fragment chunks
#define VSTR 112   // 224 B rows -> 32B-aligned fragment chunks
#define SSTR 100   // 400 B rows -> 16B-aligned float4 chunks

template <int MODE>
__global__ __launch_bounds__(256) void axial_attn_kernel(
    const __bf16* __restrict__ qb, const __bf16* __restrict__ kb,
    const __bf16* __restrict__ vsrc, const float* __restrict__ lepe,
    __bf16* __restrict__ outb, float* __restrict__ outf)
{
    __shared__ __align__(32) __bf16 Qs[96 * QSTR];
    __shared__ __align__(32) __bf16 Ks[96 * QSTR];
    __shared__ __align__(32) __bf16 VT[32 * VSTR];   // transposed V: [d][seq]
    __shared__ __align__(16) float  Sm[96 * SSTR];

    const int tid  = threadIdx.x;
    const int bx   = blockIdx.x;
    const int head = bx & 7;
    const int fx   = (bx >> 3) % 96;
    const int b    = bx / (96 * 8);

    const size_t seqStride = MODE ? (size_t)WW * CCH : (size_t)CCH;
    const size_t base = MODE
        ? (((size_t)b * HH) * WW + fx) * CCH + head * HDIM
        : (((size_t)b * HH + fx) * WW) * CCH + head * HDIM;

    for (int idx = tid; idx < 96 * 32; idx += 256) {
        const int s = idx >> 5, d = idx & 31;
        const size_t g = base + (size_t)s * seqStride + d;
        Qs[s * QSTR + d] = qb[g];
        Ks[s * QSTR + d] = kb[g];
        VT[d * VSTR + s] = vsrc[g];
    }
    __syncthreads();

    const int wave = tid >> 5;
    const int lane = tid & 31;
    const int nloc = lane & 15;
    const int mh   = (lane >> 4) * 8;

    // --- scores: 6x6 tiles, K = 32 = one WMMA each ---
    for (int t = wave; t < 36; t += 8) {
        const int mi = t / 6, ni = t % 6;
        bhalf16 a  = frag_a_lds(Qs + mi * 16 * QSTR, QSTR);
        bhalf16 bf = frag_row32(Ks + ni * 16 * QSTR, QSTR);
        floatx8 acc = {};
        acc = wmma_bf16(a, bf, acc);
#pragma unroll
        for (int r = 0; r < 8; ++r)
            Sm[(mi * 16 + mh + r) * SSTR + ni * 16 + nloc] = acc[r];
    }
    __syncthreads();

    // --- log-softmax over keys (one thread per row) ---
    if (tid < 96) {
        float* row = Sm + tid * SSTR;
        float mx = -3.4e38f;
        for (int j = 0; j < 96; ++j) mx = fmaxf(mx, row[j]);
        float sum = 0.f;
        for (int j = 0; j < 96; ++j) sum += __expf(row[j] - mx);
        const float lse = mx + __logf(sum);
        for (int j = 0; j < 96; ++j) row[j] -= lse;
    }
    __syncthreads();

    // --- OUT = logprobs @ V : 6x2 tiles, 3 K-steps of 32 ---
    for (int t = wave; t < 12; t += 8) {
        const int mi = t >> 1, ni = t & 1;
        floatx8 acc = {};
#pragma unroll
        for (int ks = 0; ks < 3; ++ks) {
            bhalf16 a  = frag_a_f32(Sm + mi * 16 * SSTR + ks * 32, SSTR);
            bhalf16 bf = frag_row32(VT + (ni * 16) * VSTR + ks * 32, VSTR);
            acc = wmma_bf16(a, bf, acc);
        }
#pragma unroll
        for (int r = 0; r < 8; ++r) {
            const int s = mi * 16 + mh + r;
            const int d = ni * 16 + nloc;
            const size_t g = base + (size_t)s * seqStride + d;
            if (MODE == 0) outb[g] = (__bf16)acc[r];
            else           outf[g] = acc[r] + lepe[g];
        }
    }
}

// ---------------------------------------------------------------------------
// Kernel 5: output projection  out = (attn+lepe) @ Wo + bo  (f32 out)
// ---------------------------------------------------------------------------
__global__ __launch_bounds__(256) void oproj_kernel(
    const float* __restrict__ attn, const __bf16* __restrict__ swzO,
    const float* __restrict__ bo, float* __restrict__ out)
{
    __shared__ __align__(32) __bf16 As[16 * ASTR];

    const int tid = threadIdx.x;
    const int m0  = blockIdx.x * 16;

    for (int idx = tid; idx < 16 * 256; idx += 256) {
        const int r = idx >> 8, c = idx & 255;
        As[r * ASTR + c] = (__bf16)attn[(size_t)(m0 + r) * CCH + c];
    }
    __syncthreads();

    const int wave = tid >> 5;
    const int lane = tid & 31;
    const int nloc = lane & 15;
    const int mh   = (lane >> 4) * 8;

    bhalf16 afr[8];
#pragma unroll
    for (int ks = 0; ks < 8; ++ks) afr[ks] = frag_a_lds(As + ks * 32, ASTR);

    for (int nt = wave; nt < 16; nt += 8) {
        floatx8 acc = {};
#pragma unroll
        for (int ks = 0; ks < 8; ++ks) {
            if (ks < 7)
                __builtin_prefetch(swzO + (size_t)((ks + 1) * 16 + nt) * 512
                                        + lane * 16, 0, 0);
            bhalf16 b = frag_b_swz(swzO + (size_t)(ks * 16 + nt) * 512);
            acc = wmma_bf16(afr[ks], b, acc);
        }
        const int col = nt * 16 + nloc;
        const float bias = bo[col];
#pragma unroll
        for (int r = 0; r < 8; ++r)
            out[(size_t)(m0 + mh + r) * CCH + col] = acc[r] + bias;
    }
}

// ---------------------------------------------------------------------------
// Host launcher
// ---------------------------------------------------------------------------
extern "C" void kernel_launch(void* const* d_in, const int* in_sizes, int n_in,
                              void* d_out, int out_size, void* d_ws, size_t ws_size,
                              hipStream_t stream) {
    const float* x    = (const float*)d_in[0];
    const float* enc  = (const float*)d_in[1];
    const float* Wq   = (const float*)d_in[2];
    const float* bq   = (const float*)d_in[3];
    const float* Wk   = (const float*)d_in[4];
    const float* bk   = (const float*)d_in[5];
    const float* Wv   = (const float*)d_in[6];
    const float* bv   = (const float*)d_in[7];
    const float* cw   = (const float*)d_in[8];
    const float* cb   = (const float*)d_in[9];
    const float* Wo   = (const float*)d_in[10];
    const float* bo   = (const float*)d_in[11];
    float* out = (float*)d_out;

    char* ws = (char*)d_ws;
    const size_t bf16Bytes = NELT * sizeof(__bf16);   // 37,748,736
    const size_t f32Bytes  = NELT * sizeof(float);    // 75,497,472
    __bf16* qb     = (__bf16*)(ws);
    __bf16* kb     = (__bf16*)(ws + bf16Bytes);
    __bf16* vb     = (__bf16*)(ws + 2 * bf16Bytes);
    __bf16* v1b    = (__bf16*)(ws + 3 * bf16Bytes);
    float*  lepe   = (float*)(ws + 4 * bf16Bytes);
    float*  att2   = (float*)(ws + 4 * bf16Bytes + f32Bytes);
    __bf16* swzQKV = (__bf16*)(ws + 4 * bf16Bytes + 2 * f32Bytes);
    __bf16* swzO   = (__bf16*)(ws + 4 * bf16Bytes + 2 * f32Bytes + 393216);

    const int mtiles = NTOK / 16;                 // 4608
    const int attnBlocks = BB * 96 * NHEAD;       // 6144
    const int convBlocks = (int)((NELT + 255) / 256);

    wprep_kernel<<<1024, 256, 0, stream>>>(Wq, Wk, Wv, Wo, swzQKV, swzO);
    qkv_kernel<<<mtiles, 256, 0, stream>>>(x, enc, swzQKV, bq, bk, bv,
                                           qb, kb, vb);
    lepe_kernel<<<convBlocks, 256, 0, stream>>>(vb, cw, cb, lepe);
    axial_attn_kernel<0><<<attnBlocks, 256, 0, stream>>>(qb, kb, vb,
                                                         nullptr, v1b, nullptr);
    axial_attn_kernel<1><<<attnBlocks, 256, 0, stream>>>(qb, kb, v1b,
                                                         lepe, nullptr, att2);
    oproj_kernel<<<mtiles, 256, 0, stream>>>(att2, swzO, bo, out);
}
